// LinearAttention_71622874628681
// MI455X (gfx1250) — matrix operand order repaired and checked by reading, not compiled, not run on hardware
//
#include <hip/hip_runtime.h>
#include <hip/hip_bf16.h>

typedef __attribute__((ext_vector_type(16))) _Float16 v16h;
typedef __attribute__((ext_vector_type(8)))  _Float16 v8h;
typedef __attribute__((ext_vector_type(8)))  float    v8f;

// Problem constants (match reference setup_inputs)
#define BB   2
#define TT   2048
#define DD   512
#define HH   16
#define DHH  32
#define MM   (BB * TT)     // 4096 rows
#define KK   DD            // 512 reduction dim
#define N3   (3 * DD)      // 1536 qkv cols

// ---------------------------------------------------------------------------
// fp32 -> f16 conversion of x, w_qkv, w_out
// ---------------------------------------------------------------------------
__global__ void cvt_f16_kernel(const float* __restrict__ x,
                               const float* __restrict__ wqkv,
                               const float* __restrict__ wout,
                               _Float16* __restrict__ xh,
                               _Float16* __restrict__ wqh,
                               _Float16* __restrict__ woh) {
    const int n_x  = MM * KK;
    const int n_wq = N3 * KK;
    const int n_wo = DD * DD;
    const int total = n_x + n_wq + n_wo;
    for (int i = blockIdx.x * blockDim.x + threadIdx.x; i < total;
         i += gridDim.x * blockDim.x) {
        if (i < n_x) {
            xh[i] = (_Float16)x[i];
        } else if (i < n_x + n_wq) {
            int j = i - n_x;
            wqh[j] = (_Float16)wqkv[j];
        } else {
            int j = i - n_x - n_wq;
            woh[j] = (_Float16)wout[j];
        }
    }
}

// ---------------------------------------------------------------------------
// WMMA fragment helpers (16x16x32 f16; A/B in this code's contiguous k order, C/D per ISA 7.12.2)
//  A (16x32): lane<16: row=lane, K 0..15; lane>=16: row=lane-16,
//             K 16..31 contiguous.  Loaded as two b128s at +khalf and +khalf+8.
//  B (32x16): lane<16: col=lane, K=0..15 contiguous; lane>=16: col=lane-16,
//             K=16..31 contiguous (same k order as A).  Two contiguous b128s.
//  C/D (16x16 f32): lane<16: col=lane, VGPR r = row r; lane>=16: row 8+r.
// ---------------------------------------------------------------------------
__device__ __forceinline__ v16h load_fragA(const _Float16* p) {
    union U { v16h v; v8h h[2]; } u;
    u.h[0] = *reinterpret_cast<const v8h*>(p);        // K +0..7 of lane's set
    u.h[1] = *reinterpret_cast<const v8h*>(p + 8);    // K +8..15 of lane's set
    return u.v;
}

__device__ __forceinline__ v16h load_fragB(const _Float16* p) {
    union U { v16h v; v8h h[2]; } u;
    u.h[0] = *reinterpret_cast<const v8h*>(p);        // K +0..7
    u.h[1] = *reinterpret_cast<const v8h*>(p + 8);    // K +8..15
    return u.v;
}

__device__ __forceinline__ v8f wmma_f16(v16h a, v16h b, v8f c) {
    return __builtin_amdgcn_wmma_f32_16x16x32_f16(
        false, a, false, b, (short)0, c, false, false);
}

// LDS byte offset of a generic pointer into a __shared__ array
__device__ __forceinline__ unsigned lds_off(const void* p) {
    return (unsigned)(unsigned long long)
        (__attribute__((address_space(3))) const void*)p;
}

// Async DMA of 16 bytes global -> LDS (per active lane); tracked by ASYNCcnt.
__device__ __forceinline__ void async_g2l_b128(unsigned lds_byte_off,
                                               const void* gptr) {
    asm volatile("global_load_async_to_lds_b128 %0, %1, off"
                 :: "v"(lds_byte_off),
                    "v"((unsigned long long)gptr)
                 : "memory");
}

__device__ __forceinline__ void wait_asynccnt0() {
    asm volatile("s_wait_asynccnt 0x0" ::: "memory");
}

__device__ __forceinline__ void wait_dscnt0() {
    asm volatile("s_wait_dscnt 0x0" ::: "memory");
}

// Compute a 32x32 output tile: A rows [m0,m0+32), W rows [n0,n0+32), lda=ldb=K
__device__ __forceinline__ void wave_tile_32x32(const _Float16* __restrict__ A,
                                                const _Float16* __restrict__ B,
                                                int Kdim,
                                                v8f& c00, v8f& c01,
                                                v8f& c10, v8f& c11) {
    const int lane = threadIdx.x & 31;
    const int rl   = lane & 15;
    const int hi   = lane >> 4;
    const int ka   = hi ? 16 : 0;   // A k-offset base (contiguous pattern)
    const int kb   = hi ? 16 : 0;   // B k-offset base (contiguous pattern)

    const _Float16* pa0 = A + (size_t)rl * Kdim + ka;
    const _Float16* pa1 = A + (size_t)(rl + 16) * Kdim + ka;
    const _Float16* pb0 = B + (size_t)rl * Kdim + kb;
    const _Float16* pb1 = B + (size_t)(rl + 16) * Kdim + kb;

    for (int kt = 0; kt < Kdim; kt += 32) {
        v16h a0 = load_fragA(pa0 + kt);
        v16h a1 = load_fragA(pa1 + kt);
        v16h b0 = load_fragB(pb0 + kt);
        v16h b1 = load_fragB(pb1 + kt);
        c00 = wmma_f16(a0, b0, c00);
        c01 = wmma_f16(a0, b1, c01);
        c10 = wmma_f16(a1, b0, c10);
        c11 = wmma_f16(a1, b1, c11);
    }
}

__device__ __forceinline__ float phi_elu1(float v) {
    return v > 0.0f ? v + 1.0f : __expf(v);
}

// ---------------------------------------------------------------------------
// QKV GEMM: qkv[m][n] = sum_k xh[m][k] * wqh[n][k]   (M=4096, N=1536, K=512)
// Epilogue: phi on q,k; scatter q/k/v as f16 into [B*H][T][Dh].
// ---------------------------------------------------------------------------
__global__ void qkv_gemm_kernel(const _Float16* __restrict__ xh,
                                const _Float16* __restrict__ wqh,
                                _Float16* __restrict__ qh,
                                _Float16* __restrict__ kh,
                                _Float16* __restrict__ vh) {
    const int m0 = blockIdx.x * 32;
    const int n0 = blockIdx.y * 32;

    v8f c00 = {}, c01 = {}, c10 = {}, c11 = {};
    wave_tile_32x32(xh + (size_t)m0 * KK, wqh + (size_t)n0 * KK, KK,
                    c00, c01, c10, c11);

    const int lane = threadIdx.x & 31;
    const int nl   = lane & 15;
    const int mhi  = (lane >> 4) << 3;

    const v8f* accs[4] = {&c00, &c01, &c10, &c11};
#pragma unroll
    for (int s = 0; s < 4; ++s) {
        const int mb = m0 + ((s >> 1) << 4);
        const int nb = n0 + ((s & 1) << 4);
        v8f acc = *accs[s];
#pragma unroll
        for (int r = 0; r < 8; ++r) {
            const int m = mb + mhi + r;
            const int n = nb + nl;
            const float val = acc[r];
            const int b = m >> 11;
            const int t = m & (TT - 1);
            const int which = n >> 9;       // 0:q 1:k 2:v
            const int d = n & (DD - 1);
            const int h = d >> 5;
            const int dh = d & (DHH - 1);
            const size_t idx = (((size_t)(b * HH + h)) * TT + t) * DHH + dh;
            if (which == 0)       qh[idx] = (_Float16)phi_elu1(val);
            else if (which == 1)  kh[idx] = (_Float16)phi_elu1(val);
            else                  vh[idx] = (_Float16)val;
        }
    }
}

// ---------------------------------------------------------------------------
// Chunked causal linear attention, chunk C=32, one wave per (b,h).
//   P        = Q_c K_c^T              (WMMA, causal-masked through LDS)
//   num      = Q_c S_prev + (P.M) V_c (WMMA)
//   den[t]   = q_t.ksum_prev + rowsum(P.M)[t]   (fp32 VALU)
//   S       += K_c^T V_c              (WMMA, fp32 accumulators held in D tiles)
// K and V chunks are staged into LDS via GLOBAL_LOAD_ASYNC_TO_LDS_B128
// (ASYNCcnt), so the strided column accesses for K^T / V frags hit LDS.
// ---------------------------------------------------------------------------
__global__ void linattn_chunk_kernel(const _Float16* __restrict__ qh,
                                     const _Float16* __restrict__ kh,
                                     const _Float16* __restrict__ vh,
                                     _Float16* __restrict__ att) {
    __shared__ _Float16 PM[32 * 32];    // masked P, row-major [t][s]
    __shared__ _Float16 Ksh[32 * 32];   // K chunk, row-major [s][dh]
    __shared__ _Float16 Vsh[32 * 32];   // V chunk, row-major [s][dh]

    const int bh   = blockIdx.x;        // 0..31
    const int lane = threadIdx.x & 31;
    const int b    = bh >> 4;
    const int h    = bh & (HH - 1);
    const int rl   = lane & 15;
    const int hi   = lane >> 4;         // 0 / 1
    const int ka   = hi ? 16 : 0;       // A-pattern base
    const int kb   = hi ? 16 : 0;       // B-pattern base

    const _Float16* qp = qh + (size_t)bh * TT * DHH;
    const _Float16* kp = kh + (size_t)bh * TT * DHH;
    const _Float16* vp = vh + (size_t)bh * TT * DHH;
    _Float16* ap = att + (size_t)b * TT * DD + (size_t)h * DHH;

    const unsigned koff = lds_off(&Ksh[lane * DHH]);  // lane's row in LDS
    const unsigned voff = lds_off(&Vsh[lane * DHH]);

    v8f s00 = {}, s01 = {}, s10 = {}, s11 = {};
    float ksum_l = 0.0f;                // ksum[dh = lane]

    for (int c0 = 0; c0 < TT; c0 += 32) {
        // ---- stage K and V chunks into LDS (each lane DMAs its own row).
        // DS and ASYNC counters are mutually unordered: drain DS reads of the
        // previous chunk before overwriting the staging buffers.
        wait_dscnt0();
        {
            const _Float16* kg = kp + (size_t)(c0 + lane) * DHH;
            const _Float16* vg = vp + (size_t)(c0 + lane) * DHH;
#pragma unroll
            for (int i = 0; i < 4; ++i) {       // 4 x b128 = 64B row
                async_g2l_b128(koff + i * 16, kg + i * 8);
                async_g2l_b128(voff + i * 16, vg + i * 8);
            }
        }

        // ---- Q A-frags (contiguous rows, straight from global)
        const _Float16* qrow = qp + (size_t)(c0 + rl) * DHH;
        v16h qa0 = load_fragA(qrow + ka);
        v16h qa1 = load_fragA(qrow + 16 * DHH + ka);

        // ---- inter-chunk denominator: q_t . ksum_prev (fp32)
        float d0 = 0.0f, d1 = 0.0f;
#pragma unroll
        for (int e = 0; e < 16; ++e) {
            const int idx = e + (hi ? 16 : 0);
            const float ks = __shfl(ksum_l, idx, 32);
            d0 += (float)qa0[e] * ks;
            d1 += (float)qa1[e] * ks;
        }
        d0 += __shfl_xor(d0, 16, 32);   // den row rl      (lanes rl, rl+16)
        d1 += __shfl_xor(d1, 16, 32);   // den row 16+rl

        // ---- repack previous S (fp32 D tiles) into f16 B-frags
        v16h Sb0, Sb1;
#pragma unroll
        for (int e = 0; e < 8; ++e) {
            float a  = s00[e], bb = s10[e], cc = s01[e], dd = s11[e];
            float xa = __shfl_xor(a, 16, 32);
            float xb = __shfl_xor(bb, 16, 32);
            float xc = __shfl_xor(cc, 16, 32);
            float xd = __shfl_xor(dd, 16, 32);
            Sb0[e]     = (_Float16)(hi ? xb : a);   // k = base+e   rows
            Sb0[e + 8] = (_Float16)(hi ? bb : xa);  // k = base+8+e rows
            Sb1[e]     = (_Float16)(hi ? xd : cc);
            Sb1[e + 8] = (_Float16)(hi ? dd : xc);
        }

        // ---- inter-chunk numerator: Q @ S_prev
        v8f z = {};
        v8f n00 = wmma_f16(qa0, Sb0, z);
        v8f n01 = wmma_f16(qa0, Sb1, z);
        v8f n10 = wmma_f16(qa1, Sb0, z);
        v8f n11 = wmma_f16(qa1, Sb1, z);

        // ---- K,V chunk DMA complete
        wait_asynccnt0();

        // ---- K B-frags for P = Q K^T (columns of K^T = rows of K), from LDS
        v16h kb0 = load_fragB(&Ksh[rl * DHH + kb]);          // s-cols 0..15
        v16h kb1 = load_fragB(&Ksh[(16 + rl) * DHH + kb]);   // s-cols 16..31

        v8f p00 = wmma_f16(qa0, kb0, z);
        v8f p01 = wmma_f16(qa0, kb1, z);
        v8f p10 = wmma_f16(qa1, kb0, z);
        v8f p11 = wmma_f16(qa1, kb1, z);

        // ---- causal mask P, write f16 to LDS row-major [t][s]
        const int sc0 = rl, sc1 = 16 + rl;
#pragma unroll
        for (int r = 0; r < 8; ++r) {
            const int t0 = r + 8 * hi;       // rows of p00/p01
            const int t1 = 16 + t0;          // rows of p10/p11
            PM[t0 * 32 + sc0] = (sc0 <= t0) ? (_Float16)p00[r] : (_Float16)0.0f;
            PM[t0 * 32 + sc1] = (sc1 <= t0) ? (_Float16)p01[r] : (_Float16)0.0f;
            PM[t1 * 32 + sc0] = (sc0 <= t1) ? (_Float16)p10[r] : (_Float16)0.0f;
            PM[t1 * 32 + sc1] = (sc1 <= t1) ? (_Float16)p11[r] : (_Float16)0.0f;
        }
        // DS ops are in-order within a wave; single-wave block needs no barrier.

        // ---- intra denominator: row sums of masked P (lane = row)
        float di = 0.0f;
        {
            const _Float16* prow = &PM[lane * 32];
#pragma unroll
            for (int e = 0; e < 32; ++e) di += (float)prow[e];
        }

        // ---- intra numerator: (P.M) @ V   (V B-frags from LDS columns)
        v16h pa0 = load_fragA(&PM[rl * 32 + ka]);
        v16h pa1 = load_fragA(&PM[(16 + rl) * 32 + ka]);
        v16h vb0, vb1;                   // col j, k = s
#pragma unroll
        for (int e = 0; e < 16; ++e) {
            const int s = (hi ? 16 : 0) + e;
            vb0[e] = Vsh[s * DHH + rl];
            vb1[e] = Vsh[s * DHH + 16 + rl];
        }
        n00 = wmma_f16(pa0, vb0, n00);
        n01 = wmma_f16(pa0, vb1, n01);
        n10 = wmma_f16(pa1, vb0, n10);
        n11 = wmma_f16(pa1, vb1, n11);

        // ---- state update: S += K^T @ V  (A = K^T: row dh, k = s, from LDS)
        v16h kt0, kt1;
#pragma unroll
        for (int e = 0; e < 16; ++e) {
            const int s = e + (hi ? 16 : 0);
            kt0[e] = Ksh[s * DHH + rl];
            kt1[e] = Ksh[s * DHH + 16 + rl];
        }
        s00 = wmma_f16(kt0, vb0, s00);
        s01 = wmma_f16(kt0, vb1, s01);
        s10 = wmma_f16(kt1, vb0, s10);
        s11 = wmma_f16(kt1, vb1, s11);

        // ---- ksum update (fp32): ksum[dh] += sum_s K[s][dh]
        float ps0 = 0.0f, ps1 = 0.0f;
#pragma unroll
        for (int e = 0; e < 16; ++e) {
            ps0 += (float)kt0[e];
            ps1 += (float)kt1[e];
        }
        ps0 += __shfl_xor(ps0, 16, 32);  // full sum for dh = rl
        ps1 += __shfl_xor(ps1, 16, 32);  // full sum for dh = 16+rl
        ksum_l += hi ? ps1 : ps0;

        // ---- write outputs: att[t][h*32 + j] = num / max(den, 1e-6)
#pragma unroll
        for (int r = 0; r < 8; ++r) {
            const int t0 = r + 8 * hi;       // rows of n00/n01
            const int t1 = 16 + t0;
            float den0 = __shfl(d0, t0, 32) + __shfl(di, t0, 32);
            float den1 = __shfl(d1, t0, 32) + __shfl(di, t1, 32);
            den0 = fmaxf(den0, 1e-6f);
            den1 = fmaxf(den1, 1e-6f);
            const size_t r0 = (size_t)(c0 + t0) * DD;
            const size_t r1 = (size_t)(c0 + t1) * DD;
            ap[r0 + rl]      = (_Float16)(n00[r] / den0);
            ap[r0 + 16 + rl] = (_Float16)(n01[r] / den0);
            ap[r1 + rl]      = (_Float16)(n10[r] / den1);
            ap[r1 + 16 + rl] = (_Float16)(n11[r] / den1);
        }
    }
}

// ---------------------------------------------------------------------------
// Output GEMM: out[m][n] = sum_d att[m][d] * woh[n][d]  (M=4096, N=K=512)
// ---------------------------------------------------------------------------
__global__ void out_gemm_kernel(const _Float16* __restrict__ att,
                                const _Float16* __restrict__ woh,
                                float* __restrict__ out) {
    const int m0 = blockIdx.x * 32;
    const int n0 = blockIdx.y * 32;

    v8f c00 = {}, c01 = {}, c10 = {}, c11 = {};
    wave_tile_32x32(att + (size_t)m0 * KK, woh + (size_t)n0 * KK, KK,
                    c00, c01, c10, c11);

    const int lane = threadIdx.x & 31;
    const int nl   = lane & 15;
    const int mhi  = (lane >> 4) << 3;

    const v8f* accs[4] = {&c00, &c01, &c10, &c11};
#pragma unroll
    for (int s = 0; s < 4; ++s) {
        const int mb = m0 + ((s >> 1) << 4);
        const int nb = n0 + ((s & 1) << 4);
        v8f acc = *accs[s];
#pragma unroll
        for (int r = 0; r < 8; ++r) {
            const int m = mb + mhi + r;
            const int n = nb + nl;
            out[(size_t)m * DD + n] = acc[r];
        }
    }
}

// ---------------------------------------------------------------------------
// Launch
// ---------------------------------------------------------------------------
extern "C" void kernel_launch(void* const* d_in, const int* in_sizes, int n_in,
                              void* d_out, int out_size, void* d_ws, size_t ws_size,
                              hipStream_t stream) {
    (void)in_sizes; (void)n_in; (void)out_size; (void)ws_size;

    const float* x    = (const float*)d_in[0];
    const float* wqkv = (const float*)d_in[1];
    const float* wout = (const float*)d_in[2];
    float* out = (float*)d_out;

    char* ws = (char*)d_ws;
    size_t off = 0;
    _Float16* xh  = (_Float16*)(ws + off); off += (size_t)MM * KK * sizeof(_Float16);
    _Float16* wqh = (_Float16*)(ws + off); off += (size_t)N3 * KK * sizeof(_Float16);
    _Float16* woh = (_Float16*)(ws + off); off += (size_t)DD * DD * sizeof(_Float16);
    _Float16* qh  = (_Float16*)(ws + off); off += (size_t)MM * DD * sizeof(_Float16);
    _Float16* kh  = (_Float16*)(ws + off); off += (size_t)MM * DD * sizeof(_Float16);
    _Float16* vh  = (_Float16*)(ws + off); off += (size_t)MM * DD * sizeof(_Float16);
    _Float16* att = (_Float16*)(ws + off); off += (size_t)MM * DD * sizeof(_Float16);

    // 1) fp32 -> f16 conversion of GEMM operands
    {
        const int total = MM * KK + N3 * KK + DD * DD;
        const int blk = 256;
        int grid = (total + blk * 4 - 1) / (blk * 4);
        if (grid > 4096) grid = 4096;
        cvt_f16_kernel<<<grid, blk, 0, stream>>>(x, wqkv, wout, xh, wqh, woh);
    }

    // 2) QKV projection + phi
    {
        dim3 grid(MM / 32, N3 / 32);
        qkv_gemm_kernel<<<grid, 32, 0, stream>>>(xh, wqh, qh, kh, vh);
    }

    // 3) chunked causal linear attention (WMMA + async LDS staging)
    linattn_chunk_kernel<<<BB * HH, 32, 0, stream>>>(qh, kh, vh, att);

    // 4) output projection
    {
        dim3 grid(MM / 32, DD / 32);
        out_gemm_kernel<<<grid, 32, 0, stream>>>(att, woh, out);
    }
}
